// UniADTransformer_80144089744040
// MI455X (gfx1250) — matrix operand order repaired and checked by reading, not compiled, not run on hardware
//
#include <hip/hip_runtime.h>

// ---------------------------------------------------------------------------
// UniAD-style transformer for MI455X (gfx1250), bf16 WMMA + f32 accumulate.
#define FDIM   272
#define HD     256
#define NHEADS 8
#define HDH    32          // head dim
#define NENC   4
#define NDEC   4
#define FFD    1024
#define FS     14
#define NTOK   196         // FS*FS
#define BATCH  64
#define MROWS  (BATCH*NTOK)   // 12544 (multiple of 16)
#define KPAD_IN 288           // 272 padded to multiple of 32
#define MPAD   208            // 196 padded to 13*16 (row tiles)
#define MTILES 13
#define NPADK  224            // padded key length, multiple of 32
#define LN_EPS 1e-5f

typedef __attribute__((ext_vector_type(16))) __bf16 bf16x16;
typedef __attribute__((ext_vector_type(8)))  float  floatx8;
typedef __attribute__((address_space(3))) unsigned short lds_us;

union Frag {
    bf16x16 v;
    unsigned int u[8];
};

__device__ __forceinline__ unsigned short f2bf(float f) {
    unsigned int u = __float_as_uint(f);
    u += 0x7fffu + ((u >> 16) & 1u);          // round-to-nearest-even
    return (unsigned short)(u >> 16);
}

// ---------------------------------------------------------------------------
// Weight cast + transpose: src f32 [K, Nc] row-major -> dst bf16 [Nc, Kpad]
__global__ void k_cast_wT(const float* __restrict__ w, unsigned short* __restrict__ wt,
                          int K, int Nc, int Kpad) {
    int idx = blockIdx.x * blockDim.x + threadIdx.x;
    if (idx >= Nc * Kpad) return;
    int n = idx / Kpad, k = idx % Kpad;
    wt[idx] = (k < K) ? f2bf(w[(size_t)k * Nc + n]) : (unsigned short)0;
}

// tokens f32 [M, K] -> bf16 [M, Kpad], zero padded
__global__ void k_cast_pad(const float* __restrict__ src, unsigned short* __restrict__ dst,
                           int M, int K, int Kpad) {
    int idx = blockIdx.x * blockDim.x + threadIdx.x;
    if (idx >= M * Kpad) return;
    int m = idx / Kpad, k = idx % Kpad;
    dst[idx] = (k < K) ? f2bf(src[(size_t)m * K + k]) : (unsigned short)0;
}

__global__ void k_addpos(float* __restrict__ x, const float* __restrict__ pos) {
    int idx = blockIdx.x * blockDim.x + threadIdx.x;
    if (idx >= MROWS * HD) return;
    x[idx] += pos[idx % (NTOK * HD)];
}

__global__ void k_buildq(const float* __restrict__ qi, const float* __restrict__ pos,
                         float* __restrict__ q) {
    int idx = blockIdx.x * blockDim.x + threadIdx.x;
    if (idx >= MROWS * HD) return;
    int rc = idx % (NTOK * HD);
    q[idx] = qi[rc] + pos[rc];
}

// LayerNorm over HD=256, one row per block (256 threads).
__global__ void k_layernorm(const float* __restrict__ x, const float* __restrict__ g,
                            const float* __restrict__ b, float* __restrict__ outf,
                            unsigned short* __restrict__ outb) {
    int row = blockIdx.x, c = threadIdx.x;
    __shared__ float red[HD];
    float v = x[(size_t)row * HD + c];
    red[c] = v; __syncthreads();
    for (int s = HD / 2; s > 0; s >>= 1) { if (c < s) red[c] += red[c + s]; __syncthreads(); }
    float mean = red[0] * (1.0f / HD);
    __syncthreads();
    float d = v - mean;
    red[c] = d * d; __syncthreads();
    for (int s = HD / 2; s > 0; s >>= 1) { if (c < s) red[c] += red[c + s]; __syncthreads(); }
    float var = red[0] * (1.0f / HD);
    float y = d * rsqrtf(var + LN_EPS) * g[c] + b[c];
    if (outf) outf[(size_t)row * HD + c] = y;
    if (outb) outb[(size_t)row * HD + c] = f2bf(y);
}

// ---------------------------------------------------------------------------
// Generic bf16 WMMA GEMM, branch-free: each wave computes 16 x (16*NSUB).
// A tile (16x32 bf16, 1KB) is staged global->LDS with async-LDS loads
// (double buffered, ASYNCcnt), A fragments are then built from LDS.
// Requires M%16==0, K%32==0; NSUB in {1..4} covers Nc%64 remainders.
template <int NSUB>
__global__ void k_gemm(const unsigned short* __restrict__ A, int lda,
                       const unsigned short* __restrict__ Wt,
                       const float* __restrict__ bias,
                       int Nc, int K, int ncol0,
                       const float* __restrict__ resid,
                       float* __restrict__ outf,
                       unsigned short* __restrict__ outb,
                       int act) {
    __shared__ __align__(16) unsigned short Abuf[2][16 * 32];
    int lane   = threadIdx.x;
    int mtile  = blockIdx.y;
    int n0base = ncol0 + blockIdx.x * (16 * NSUB);
    int asel   = (lane < 16) ? 0 : 8;    // A: lo lanes K 0..7/16..23, hi 8..15/24..31
    int bsel   = (lane < 16) ? 0 : 8;    // B: lo lanes K 0..7/16..23, hi 8..15/24..31

    const unsigned short* Atile = A + (size_t)(mtile * 16) * lda;

    // async stage lane mapping: lane -> (row, 16B chunk); 2 instrs move 1KB tile
    int srow = lane >> 1;
    int sch  = lane & 1;
    auto stage = [&](int k0, int bufi) {
        const unsigned short* g0 = Atile + (size_t)srow * lda + k0 + sch * 8;
        lds_us* l0 = (lds_us*)&Abuf[bufi][srow * 32 + sch * 8];
        asm volatile("global_load_async_to_lds_b128 %0, %1, off"
                     :: "v"(l0), "v"(g0) : "memory");
        const unsigned short* g1 = g0 + 16;
        lds_us* l1 = (lds_us*)&Abuf[bufi][srow * 32 + sch * 8 + 16];
        asm volatile("global_load_async_to_lds_b128 %0, %1, off"
                     :: "v"(l1), "v"(g1) : "memory");
    };

    floatx8 z8 = {0.f,0.f,0.f,0.f,0.f,0.f,0.f,0.f};
    floatx8 acc[NSUB];
    #pragma unroll
    for (int j = 0; j < NSUB; j++) acc[j] = z8;

    stage(0, 0);
    asm volatile("s_wait_asynccnt 0" ::: "memory");
    int cur = 0;

    for (int k0 = 0; k0 < K; k0 += 32) {
        if (k0 + 32 < K) {
            stage(k0 + 32, cur ^ 1);                       // prefetch next tile -> LDS
            __builtin_prefetch(Wt + (size_t)(n0base + (lane & 15)) * K + k0 + 32, 0, 1);
        }
        // A fragment from LDS (ds_load_b128 x2 per lane)
        Frag a;
        {
            const unsigned int* lp =
                (const unsigned int*)&Abuf[cur][(lane & 15) * 32 + asel];
            const unsigned int* lp2 =
                (const unsigned int*)&Abuf[cur][(lane & 15) * 32 + asel + 16];
            #pragma unroll
            for (int j = 0; j < 4; j++) { a.u[j] = lp[j]; a.u[4 + j] = lp2[j]; }
        }
        // all B fragments first (lets loads batch in one clause), then WMMAs
        Frag bfr[NSUB];
        #pragma unroll
        for (int j = 0; j < NSUB; j++) {
            int col = n0base + j * 16 + (lane & 15);
            const unsigned int* bp = (const unsigned int*)(Wt + (size_t)col * K + k0 + bsel);
            #pragma unroll
            for (int t = 0; t < 8; t++) bfr[j].u[t] = bp[t < 4 ? t : t + 4];
        }
        #pragma unroll
        for (int j = 0; j < NSUB; j++)
            acc[j] = __builtin_amdgcn_wmma_f32_16x16x32_bf16(
                false, a.v, false, bfr[j].v, (short)0, acc[j], false, false);

        asm volatile("s_wait_asynccnt 0" ::: "memory");    // next tile resident
        cur ^= 1;
    }

    // epilogue: bias (+GELU) (+residual), fp32 and/or bf16 stores
    #pragma unroll
    for (int j = 0; j < NSUB; j++) {
        int col = n0base + j * 16 + (lane & 15);
        float bv = bias ? bias[col] : 0.f;
        #pragma unroll
        for (int r = 0; r < 8; r++) {
            int orow = mtile * 16 + r + ((lane < 16) ? 0 : 8);
            float v = acc[j][r] + bv;
            if (act == 1) v = 0.5f * v * (1.f + erff(v * 0.70710678f)); // exact GELU
            size_t o = (size_t)orow * Nc + col;
            if (resid) v += resid[o];
            if (outf) outf[o] = v;
            if (outb) outb[o] = f2bf(v);
        }
    }
}

// ---------------------------------------------------------------------------
// V transpose: Vb bf16 [B*N, HD] -> Vt bf16 [B*NH, HDH, NPADK], zero pad t>=196
__global__ void k_transpose_v(const unsigned short* __restrict__ Vb,
                              unsigned short* __restrict__ Vt) {
    int idx = blockIdx.x * blockDim.x + threadIdx.x;
    int total = BATCH * NHEADS * HDH * NPADK;
    if (idx >= total) return;
    int t = idx % NPADK; int rest = idx / NPADK;
    int d = rest % HDH; rest /= HDH;
    int h = rest % NHEADS; int b = rest / NHEADS;
    unsigned short v = 0;
    if (t < NTOK) v = Vb[((size_t)(b * NTOK + t)) * HD + h * HDH + d];
    Vt[idx] = v;
}

// ---------------------------------------------------------------------------
// Fused Q@K^T (WMMA, K=32 one step) + neighbor mask + softmax -> bf16 probs.
// One wave per (b, h, mtile). probs layout: [B*NH, MPAD, NPADK].
__global__ void k_attn_scores(const unsigned short* __restrict__ Qb,
                              const unsigned short* __restrict__ Kb,
                              unsigned short* __restrict__ probs) {
    __shared__ float sc[16][MPAD];
    int lane  = threadIdx.x;
    int mtile = blockIdx.x % MTILES;
    int bh    = blockIdx.x / MTILES;
    int b = bh / NHEADS, h = bh % NHEADS;
    int qtok = mtile * 16 + (lane & 15);
    int asel = (lane < 16) ? 0 : 8;
    int bsel = (lane < 16) ? 0 : 8;

    Frag a;
    if (qtok < NTOK) {
        const unsigned int* ap =
            (const unsigned int*)(Qb + ((size_t)(b * NTOK + qtok)) * HD + h * HDH + asel);
        #pragma unroll
        for (int j = 0; j < 4; j++) { a.u[j] = ap[j]; a.u[4 + j] = ap[8 + j]; }
    } else {
        #pragma unroll
        for (int j = 0; j < 8; j++) a.u[j] = 0;
    }

    for (int nt = 0; nt < MTILES; nt++) {
        int ktok = nt * 16 + (lane & 15);
        Frag bf;
        if (ktok < NTOK) {
            const unsigned int* bp =
                (const unsigned int*)(Kb + ((size_t)(b * NTOK + ktok)) * HD + h * HDH + bsel);
            #pragma unroll
            for (int t = 0; t < 8; t++) bf.u[t] = bp[t < 4 ? t : t + 4];
        } else {
            #pragma unroll
            for (int t = 0; t < 8; t++) bf.u[t] = 0;
        }
        floatx8 c = {0.f,0.f,0.f,0.f,0.f,0.f,0.f,0.f};
        c = __builtin_amdgcn_wmma_f32_16x16x32_bf16(false, a.v, false, bf.v,
                                                    (short)0, c, false, false);
        int kc = nt * 16 + (lane & 15);
        #pragma unroll
        for (int r = 0; r < 8; r++) {
            int qr = r + ((lane < 16) ? 0 : 8);
            int qa = mtile * 16 + qr;                  // absolute query token (may be >=196)
            float v = c[r] * 0.17677669529663689f;     // 1/sqrt(32)
            if (kc >= NTOK) v = -1e30f;
            else {
                int dr = qa / FS - kc / FS; if (dr < 0) dr = -dr;
                int dc = qa % FS - kc % FS; if (dc < 0) dc = -dc;
                int cheb = dr > dc ? dr : dc;
                if (cheb <= 3) v = -1e30f;             // masked_fill(-inf) on neighbors
            }
            sc[qr][kc] = v;
        }
    }
    __syncthreads();

    if (lane < 16) {
        int row = lane;
        float m = -1e30f;
        for (int j = 0; j < MPAD; j++) m = fmaxf(m, sc[row][j]);
        float s = 0.f;
        for (int j = 0; j < MPAD; j++) { float e = expf(sc[row][j] - m); sc[row][j] = e; s += e; }
        float inv = 1.f / s;
        size_t base = ((size_t)bh * MPAD + mtile * 16 + row) * NPADK;
        for (int j = 0; j < NPADK; j++) {
            float p = (j < MPAD) ? sc[row][j] * inv : 0.f;
            probs[base + j] = f2bf(p);
        }
    }
}

// attn @ V: probs [B*NH, MPAD, NPADK] x Vt [B*NH, HDH, NPADK] -> out bf16 [B*N, HD]
__global__ void k_attn_av(const unsigned short* __restrict__ probs,
                          const unsigned short* __restrict__ Vt,
                          unsigned short* __restrict__ outb) {
    int lane  = threadIdx.x;
    int mtile = blockIdx.x % MTILES;
    int bh    = blockIdx.x / MTILES;
    int b = bh / NHEADS, h = bh % NHEADS;
    int asel = (lane < 16) ? 0 : 8;
    int bsel = (lane < 16) ? 0 : 8;

    floatx8 z8 = {0.f,0.f,0.f,0.f,0.f,0.f,0.f,0.f};
    floatx8 acc[2] = {z8, z8};

    const unsigned short* arow =
        probs + ((size_t)bh * MPAD + mtile * 16 + (lane & 15)) * NPADK;
    for (int k0 = 0; k0 < NPADK; k0 += 32) {
        Frag a;
        const unsigned int* ap  = (const unsigned int*)(arow + k0 + asel);
        const unsigned int* ap2 = (const unsigned int*)(arow + k0 + asel + 16);
        #pragma unroll
        for (int j = 0; j < 4; j++) { a.u[j] = ap[j]; a.u[4 + j] = ap2[j]; }
        Frag bfr[2];
        #pragma unroll
        for (int j = 0; j < 2; j++) {
            int d = j * 16 + (lane & 15);
            const unsigned int* bp =
                (const unsigned int*)(Vt + ((size_t)(bh * HDH + d)) * NPADK + k0 + bsel);
            #pragma unroll
            for (int t = 0; t < 8; t++) bfr[j].u[t] = bp[t < 4 ? t : t + 4];
        }
        #pragma unroll
        for (int j = 0; j < 2; j++)
            acc[j] = __builtin_amdgcn_wmma_f32_16x16x32_bf16(
                false, a.v, false, bfr[j].v, (short)0, acc[j], false, false);
    }
    #pragma unroll
    for (int j = 0; j < 2; j++) {
        int d = j * 16 + (lane & 15);
        #pragma unroll
        for (int r = 0; r < 8; r++) {
            int tok = mtile * 16 + r + ((lane < 16) ? 0 : 8);
            if (tok < NTOK)
                outb[((size_t)(b * NTOK + tok)) * HD + h * HDH + d] = f2bf(acc[j][r]);
        }
    }
}

// ===========================================================================
// Host orchestration
// ===========================================================================
namespace {

struct Ws {
    char* base; size_t cur;
    void* take(size_t bytes) {
        void* p = base + cur;
        cur += (bytes + 255) & ~(size_t)255;
        return p;
    }
};

inline void gemm(hipStream_t s, const unsigned short* A, int lda,
                 const unsigned short* Wt, const float* bias,
                 int M, int Nc, int K,
                 const float* resid, float* outf, unsigned short* outb, int act) {
    int full = Nc / 64;
    if (full)
        k_gemm<4><<<dim3(full, M / 16), 32, 0, s>>>(A, lda, Wt, bias, Nc, K, 0,
                                                    resid, outf, outb, act);
    int rem = (Nc % 64) / 16;
    int base = full * 64;
    if (rem == 1)
        k_gemm<1><<<dim3(1, M / 16), 32, 0, s>>>(A, lda, Wt, bias, Nc, K, base,
                                                 resid, outf, outb, act);
    else if (rem == 2)
        k_gemm<2><<<dim3(1, M / 16), 32, 0, s>>>(A, lda, Wt, bias, Nc, K, base,
                                                 resid, outf, outb, act);
    else if (rem == 3)
        k_gemm<3><<<dim3(1, M / 16), 32, 0, s>>>(A, lda, Wt, bias, Nc, K, base,
                                                 resid, outf, outb, act);
}

inline unsigned short* castw(hipStream_t s, Ws& ws, const float* w, int K, int Nc, int Kpad) {
    unsigned short* dst = (unsigned short*)ws.take((size_t)Nc * Kpad * 2);
    int total = Nc * Kpad;
    k_cast_wT<<<(total + 255) / 256, 256, 0, s>>>(w, dst, K, Nc, Kpad);
    return dst;
}

inline void ln(hipStream_t s, const float* x, const float* g, const float* b,
               float* outf, unsigned short* outb) {
    k_layernorm<<<MROWS, HD, 0, s>>>(x, g, b, outf, outb);
}

struct MhaW { unsigned short *wq, *wk, *wv, *wo; const float *bq, *bk, *bv, *bo; };
struct Scratch {
    unsigned short *Qb, *Kb, *Vb, *Vt, *probs, *AOb, *Hb, *Xb1, *Xb2;
};

inline void run_mha(hipStream_t s, const Scratch& sc,
                    const unsigned short* Aq, const unsigned short* Akv,
                    const MhaW& w, float* x /* residual in/out, fp32 [M,HD] */) {
    gemm(s, Aq,  HD, w.wq, w.bq, MROWS, HD, HD, nullptr, nullptr, sc.Qb, 0);
    gemm(s, Akv, HD, w.wk, w.bk, MROWS, HD, HD, nullptr, nullptr, sc.Kb, 0);
    gemm(s, Akv, HD, w.wv, w.bv, MROWS, HD, HD, nullptr, nullptr, sc.Vb, 0);
    int vt_total = BATCH * NHEADS * HDH * NPADK;
    k_transpose_v<<<(vt_total + 255) / 256, 256, 0, s>>>(sc.Vb, sc.Vt);
    k_attn_scores<<<BATCH * NHEADS * MTILES, 32, 0, s>>>(sc.Qb, sc.Kb, sc.probs);
    k_attn_av<<<BATCH * NHEADS * MTILES, 32, 0, s>>>(sc.probs, sc.Vt, sc.AOb);
    gemm(s, sc.AOb, HD, w.wo, w.bo, MROWS, HD, HD, x, x, nullptr, 0);
}

} // namespace

extern "C" void kernel_launch(void* const* d_in, const int* in_sizes, int n_in,
                              void* d_out, int out_size, void* d_ws, size_t ws_size,
                              hipStream_t stream) {
    (void)in_sizes; (void)n_in; (void)out_size; (void)ws_size;
    // Input flattening assumption: setup_inputs() dict leaves in insertion order.
    #define IN(i) ((const float*)d_in[(i)])
    const float* tokens  = IN(0);
    const float* in_w    = IN(1);  const float* in_b  = IN(2);
    const float* out_w   = IN(3);  const float* out_b = IN(4);
    const float* pos     = IN(5);
    const float* queries = IN(6);
    const int ENC0 = 7;                // 16 leaves per encoder layer
    const int DEC0 = 7 + NENC * 16;    // 26 leaves per decoder layer
    const int NRM_ENC = DEC0 + NDEC * 26;
    const int NRM_DEC = NRM_ENC + 2;
    // d_in[NRM_DEC+2] is the mask; recomputed on device from constants instead.

    Ws ws{ (char*)d_ws, 0 };

    // ---- weight casts (bf16, transposed) -------------------------------
    unsigned short* WinT  = castw(stream, ws, in_w,  FDIM, HD,  KPAD_IN);
    unsigned short* WoutT = castw(stream, ws, out_w, HD,   FDIM, HD);

    MhaW enc_attn[NENC]; unsigned short *enc_l1[NENC], *enc_l2[NENC];
    const float *enc_l1b[NENC], *enc_l2b[NENC];
    const float *enc_n1g[NENC], *enc_n1b[NENC], *enc_n2g[NENC], *enc_n2b[NENC];
    for (int i = 0; i < NENC; i++) {
        int p = ENC0 + i * 16;
        enc_attn[i].wq = castw(stream, ws, IN(p + 0), HD, HD, HD); enc_attn[i].bq = IN(p + 1);
        enc_attn[i].wk = castw(stream, ws, IN(p + 2), HD, HD, HD); enc_attn[i].bk = IN(p + 3);
        enc_attn[i].wv = castw(stream, ws, IN(p + 4), HD, HD, HD); enc_attn[i].bv = IN(p + 5);
        enc_attn[i].wo = castw(stream, ws, IN(p + 6), HD, HD, HD); enc_attn[i].bo = IN(p + 7);
        enc_l1[i] = castw(stream, ws, IN(p + 8),  HD,  FFD, HD);  enc_l1b[i] = IN(p + 9);
        enc_l2[i] = castw(stream, ws, IN(p + 10), FFD, HD,  FFD); enc_l2b[i] = IN(p + 11);
        enc_n1g[i] = IN(p + 12); enc_n1b[i] = IN(p + 13);
        enc_n2g[i] = IN(p + 14); enc_n2b[i] = IN(p + 15);
    }
    MhaW dec_self[NDEC], dec_cross[NDEC];
    unsigned short *dec_l1[NDEC], *dec_l2[NDEC];
    const float *dec_l1b[NDEC], *dec_l2b[NDEC];
    const float *dn1g[NDEC], *dn1b[NDEC], *dn2g[NDEC], *dn2b[NDEC], *dn3g[NDEC], *dn3b[NDEC];
    for (int i = 0; i < NDEC; i++) {
        int p = DEC0 + i * 26;
        dec_self[i].wq  = castw(stream, ws, IN(p + 0), HD, HD, HD); dec_self[i].bq  = IN(p + 1);
        dec_self[i].wk  = castw(stream, ws, IN(p + 2), HD, HD, HD); dec_self[i].bk  = IN(p + 3);
        dec_self[i].wv  = castw(stream, ws, IN(p + 4), HD, HD, HD); dec_self[i].bv  = IN(p + 5);
        dec_self[i].wo  = castw(stream, ws, IN(p + 6), HD, HD, HD); dec_self[i].bo  = IN(p + 7);
        dec_cross[i].wq = castw(stream, ws, IN(p + 8), HD, HD, HD); dec_cross[i].bq = IN(p + 9);
        dec_cross[i].wk = castw(stream, ws, IN(p + 10), HD, HD, HD); dec_cross[i].bk = IN(p + 11);
        dec_cross[i].wv = castw(stream, ws, IN(p + 12), HD, HD, HD); dec_cross[i].bv = IN(p + 13);
        dec_cross[i].wo = castw(stream, ws, IN(p + 14), HD, HD, HD); dec_cross[i].bo = IN(p + 15);
        dec_l1[i] = castw(stream, ws, IN(p + 16), HD,  FFD, HD);  dec_l1b[i] = IN(p + 17);
        dec_l2[i] = castw(stream, ws, IN(p + 18), FFD, HD,  FFD); dec_l2b[i] = IN(p + 19);
        dn1g[i] = IN(p + 20); dn1b[i] = IN(p + 21);
        dn2g[i] = IN(p + 22); dn2b[i] = IN(p + 23);
        dn3g[i] = IN(p + 24); dn3b[i] = IN(p + 25);
    }
    const float* ne_g = IN(NRM_ENC), *ne_b = IN(NRM_ENC + 1);
    const float* nd_g = IN(NRM_DEC), *nd_b = IN(NRM_DEC + 1);

    // ---- activation scratch --------------------------------------------
    unsigned short* TokP = (unsigned short*)ws.take((size_t)MROWS * KPAD_IN * 2);
    float* X    = (float*)ws.take((size_t)MROWS * HD * 4);   // encoder residual stream
    float* Mem  = (float*)ws.take((size_t)MROWS * HD * 4);   // layer_norm(enc out)
    float* Dq0  = (float*)ws.take((size_t)MROWS * HD * 4);
    float* Dq1  = (float*)ws.take((size_t)MROWS * HD * 4);
    Scratch sc;
    sc.Xb1   = (unsigned short*)ws.take((size_t)MROWS * HD * 2);
    sc.Xb2   = (unsigned short*)ws.take((size_t)MROWS * HD * 2);
    sc.Qb    = (unsigned short*)ws.take((size_t)MROWS * HD * 2);
    sc.Kb    = (unsigned short*)ws.take((size_t)MROWS * HD * 2);
    sc.Vb    = (unsigned short*)ws.take((size_t)MROWS * HD * 2);
    sc.Vt    = (unsigned short*)ws.take((size_t)BATCH * NHEADS * HDH * NPADK * 2);
    sc.probs = (unsigned short*)ws.take((size_t)BATCH * NHEADS * MPAD * NPADK * 2);
    sc.AOb   = (unsigned short*)ws.take((size_t)MROWS * HD * 2);
    sc.Hb    = (unsigned short*)ws.take((size_t)MROWS * FFD * 2);

    // ---- input projection + positional embedding -----------------------
    {
        int total = MROWS * KPAD_IN;
        k_cast_pad<<<(total + 255) / 256, 256, 0, stream>>>(tokens, TokP, MROWS, FDIM, KPAD_IN);
        gemm(stream, TokP, KPAD_IN, WinT, in_b, MROWS, HD, KPAD_IN, nullptr, X, nullptr, 0);
        int t2 = MROWS * HD;
        k_addpos<<<(t2 + 255) / 256, 256, 0, stream>>>(X, pos);
    }

    // ---- encoder --------------------------------------------------------
    for (int i = 0; i < NENC; i++) {
        ln(stream, X, enc_n1g[i], enc_n1b[i], nullptr, sc.Xb1);
        run_mha(stream, sc, sc.Xb1, sc.Xb1, enc_attn[i], X);
        ln(stream, X, enc_n2g[i], enc_n2b[i], nullptr, sc.Xb1);
        gemm(stream, sc.Xb1, HD,  enc_l1[i], enc_l1b[i], MROWS, FFD, HD, nullptr, nullptr, sc.Hb, 1);
        gemm(stream, sc.Hb,  FFD, enc_l2[i], enc_l2b[i], MROWS, HD, FFD, X, X, nullptr, 0);
    }
    ln(stream, X, ne_g, ne_b, Mem, nullptr);   // memory = layer_norm(x, norm_enc)

    // ---- decoder (layer-wise queries) -----------------------------------
    int t2 = MROWS * HD;
    for (int i = 0; i < NDEC; i++) {
        float* q   = (i & 1) ? Dq1 : Dq0;
        float* mem = (i == 0) ? Mem : ((i & 1) ? Dq0 : Dq1);
        k_buildq<<<(t2 + 255) / 256, 256, 0, stream>>>(queries + (size_t)i * NTOK * HD, pos, q);
        ln(stream, q, dn1g[i], dn1b[i], nullptr, sc.Xb1);
        run_mha(stream, sc, sc.Xb1, sc.Xb1, dec_self[i], q);
        ln(stream, q,   dn2g[i], dn2b[i], nullptr, sc.Xb1);   // q2
        ln(stream, mem, dn2g[i], dn2b[i], nullptr, sc.Xb2);   // m (same norm2 params)
        run_mha(stream, sc, sc.Xb1, sc.Xb2, dec_cross[i], q);
        ln(stream, q, dn3g[i], dn3b[i], nullptr, sc.Xb1);
        gemm(stream, sc.Xb1, HD,  dec_l1[i], dec_l1b[i], MROWS, FFD, HD, nullptr, nullptr, sc.Hb, 1);
        gemm(stream, sc.Hb,  FFD, dec_l2[i], dec_l2b[i], MROWS, HD, FFD, q, q, nullptr, 0);
    }

    // ---- final norm + output projection ---------------------------------
    float* outq = ((NDEC - 1) & 1) ? Dq1 : Dq0;
    ln(stream, outq, nd_g, nd_b, nullptr, sc.Xb1);
    gemm(stream, sc.Xb1, HD, WoutT, out_b, MROWS, FDIM, HD, nullptr, (float*)d_out, nullptr, 0);
    #undef IN
}
